// InterGate_56959856279986
// MI455X (gfx1250) — matrix operand order repaired and checked by reading, not compiled, not run on hardware
//
#include <hip/hip_runtime.h>
#include <hip/hip_bf16.h>

#define B_ 4
#define C_ 192
#define N_ 4096
#define WAVES 4

typedef __attribute__((ext_vector_type(16))) __bf16 v16bf;
typedef __attribute__((ext_vector_type(8)))  float  v8f;
typedef __attribute__((ext_vector_type(4)))  unsigned int u32x4;
typedef __attribute__((ext_vector_type(8)))  int i32x8;
typedef __attribute__((ext_vector_type(4)))  int i32x4;

union ABf {
    v16bf v;
    unsigned int u[8];
    unsigned short s[16];
};

__device__ __forceinline__ unsigned short f2bf(float x) {
    unsigned int u = __float_as_uint(x);
    u += 0x7FFFu + ((u >> 16) & 1u);       // round-to-nearest-even
    return (unsigned short)(u >> 16);
}

__device__ __forceinline__ v8f vzero() {
    v8f z;
#pragma unroll
    for (int i = 0; i < 8; ++i) z[i] = 0.f;
    return z;
}

// ---- Tensor Data Mover: 2D tile (bf16 elements) global -> LDS (ISA ch.8 D#) --
__device__ __forceinline__ void tdm_load_2d(unsigned lds_off, const void* gptr,
                                            unsigned tile_d0, unsigned tile_d1,
                                            unsigned tensor_d0, unsigned tensor_d1,
                                            unsigned long long d0_stride) {
    unsigned long long ga = (unsigned long long)(size_t)gptr;
    u32x4 g0;
    g0[0] = 1u;                                            // count=1, user desc
    g0[1] = lds_off;                                       // lds_addr [63:32]
    g0[2] = (unsigned)(ga & 0xFFFFFFFFu);                  // global_addr lo
    g0[3] = (unsigned)((ga >> 32) & 0x1FFFFFFu)            // global_addr hi (57b)
          | (2u << 30);                                    // type = 2 ("image")
    i32x8 g1;
    g1[0] = (int)(1u << 16);                               // data_size = 2 bytes
    g1[1] = (int)((tensor_d0 & 0xFFFFu) << 16);            // tensor_dim0 lo
    g1[2] = (int)((tensor_d0 >> 16) | ((tensor_d1 & 0xFFFFu) << 16));
    g1[3] = (int)((tensor_d1 >> 16) | (tile_d0 << 16));    // tile_dim0
    g1[4] = (int)(tile_d1 & 0xFFFFu);                      // tile_dim1, tile_dim2=0
    g1[5] = (int)(unsigned)(d0_stride & 0xFFFFFFFFu);      // dim0 stride lo
    g1[6] = (int)(unsigned)((d0_stride >> 32) & 0xFFFFu);  // dim0 stride hi, dim1 stride=0
    g1[7] = 0;
    i32x4 z4 = {0, 0, 0, 0};
#if __has_include(<hip/amd_detail/amd_gfx1250_TDM.h>)
    i32x8 z8 = {0, 0, 0, 0, 0, 0, 0, 0};
    __builtin_amdgcn_tensor_load_to_lds(g0, g1, z4, z4, z8, 0);
#else
    __builtin_amdgcn_tensor_load_to_lds(g0, g1, z4, z4, 0);
#endif
}

__device__ __forceinline__ unsigned lds_off_of(const void* p) {
    return (unsigned)(size_t)p;        // generic LDS ptr: low 32 bits = LDS offset
}

// ---------------- kernel 1: latp = w4 @ lat + b4, write bf16 in two layouts ----
__global__ void k_latp(const float* __restrict__ lat, const float* __restrict__ w4,
                       const float* __restrict__ b4,
                       unsigned short* __restrict__ nmaj,   // [B][N][C]
                       unsigned short* __restrict__ cmaj) { // [B][C][N]
    __shared__ float wrow[C_];
    int bo = blockIdx.x;              // b*C + o
    int b = bo / C_, o = bo % C_;
    int t = threadIdx.x;
    if (t < C_) wrow[t] = w4[o * C_ + t];
    __syncthreads();
    int n = blockIdx.y * blockDim.x + t;
    const float* latb = lat + (size_t)b * C_ * N_ + n;
    float acc = b4[o];
#pragma unroll 8
    for (int c = 0; c < C_; ++c) acc += wrow[c] * latb[(size_t)c * N_];
    unsigned short hv = f2bf(acc);
    cmaj[((size_t)b * C_ + o) * N_ + n] = hv;
    nmaj[((size_t)b * N_ + n) * C_ + o] = hv;
}

// ---------------- kernel 1b: Q = bf16(ref^T), [B][N][C] ------------------------
__global__ void k_q(const float* __restrict__ ref, unsigned short* __restrict__ q) {
    int idx = blockIdx.x * blockDim.x + threadIdx.x;
    int c = idx % C_;
    int n = (idx / C_) % N_;
    int b = idx / (C_ * N_);
    q[idx] = f2bf(ref[((size_t)b * C_ + c) * N_ + n]);
}

// ------- kernel 2: flash attention, TDM-fed LDS staging, 1 wave = 16 q rows ----
__global__ void __launch_bounds__(32 * WAVES, 1)
k_attn(const unsigned short* __restrict__ Q,     // [B][N][C] bf16
       const unsigned short* __restrict__ Knmaj, // latp [B][N][C] bf16
       const unsigned short* __restrict__ Vcmaj, // latp [B][C][N] bf16
       float* __restrict__ ctx)                  // [B][C][N] f32
{
    __shared__ __align__(16) unsigned short kbuf[2][32 * C_];  // [key][c]  12KB x2
    __shared__ __align__(16) unsigned short vbuf[2][C_ * 32];  // [c][key]  12KB x2
    __shared__ __align__(16) unsigned short pbuf[WAVES][16 * 32];

    const int lane = threadIdx.x & 31;
    const int wave = threadIdx.x >> 5;
    const int b    = blockIdx.x;
    const int q0   = (blockIdx.y * WAVES + wave) * 16;

    const int m    = lane & 15;   // A-row / B-col index within a 16-wide tile
    const int half = lane >> 4;

    const unsigned short* Kg = Knmaj + (size_t)b * N_ * C_;
    const unsigned short* Vg = Vcmaj + (size_t)b * C_ * N_;

    // --- preload Q A-fragments: 6 chunks of K=32 over C=192 (global, once)
    ABf qa[6];
    const unsigned short* Qb = Q + ((size_t)b * N_ + (q0 + m)) * C_;
#pragma unroll
    for (int j = 0; j < 6; ++j) {
#pragma unroll
        for (int v = 0; v < 8; ++v) {
            int kb = half * 16 + 2 * v;
            qa[j].u[v] = *(const unsigned int*)(Qb + j * 32 + kb);
        }
    }

    v8f acc[12];                       // O = 16 q-rows x 192 channels, 12 tiles
#pragma unroll
    for (int t2 = 0; t2 < 12; ++t2) acc[t2] = vzero();

    float rowmax[8], rowsum[8];        // online-softmax stats (row = r + 8*half)
#pragma unroll
    for (int r = 0; r < 8; ++r) { rowmax[r] = -3.0e38f; rowsum[r] = 0.f; }

    unsigned short* pb = pbuf[wave];

    // --- prologue: TDM-stage chunk 0 into buffer 0
    if (wave == 0) {
        // K tile: 32 consecutive nmaj rows = one contiguous 12KB block (1D)
        tdm_load_2d(lds_off_of(&kbuf[0][0]), Kg, 32 * C_, 1, 32 * C_, 1, 32 * C_);
        // V tile: 192 rows x 32 keys, row stride N_ elements (2D)
        tdm_load_2d(lds_off_of(&vbuf[0][0]), Vg, 32, C_, N_, C_, N_);
    }

    for (int k0 = 0; k0 < N_; k0 += 32) {
        const int buf = (k0 >> 5) & 1;
        if (wave == 0)
            asm volatile("s_wait_tensorcnt 0x0" ::: "memory");
        __syncthreads();   // publish TDM-written LDS to all waves
        if (wave == 0 && (k0 + 32) < N_) {
            tdm_load_2d(lds_off_of(&kbuf[buf ^ 1][0]),
                        Kg + (size_t)(k0 + 32) * C_, 32 * C_, 1, 32 * C_, 1, 32 * C_);
            tdm_load_2d(lds_off_of(&vbuf[buf ^ 1][0]),
                        Vg + (k0 + 32), 32, C_, N_, C_, N_);
        }
        const unsigned short* kb = kbuf[buf];
        const unsigned short* vb = vbuf[buf];

        // ---- S = Q K^T : two 16x16 f32 tiles, K-dim = 192 = 6 x 32
        v8f s0 = vzero(), s1 = vzero();
#pragma unroll
        for (int j = 0; j < 6; ++j) {
            ABf b0, b1;
            const unsigned short* p0 = kb + (size_t)m        * C_ + j * 32 + half * 16;
            const unsigned short* p1 = kb + (size_t)(16 + m) * C_ + j * 32 + half * 16;
#pragma unroll
            for (int v = 0; v < 8; ++v) {
                b0.u[v] = *(const unsigned int*)(p0 + 2 * v);
                b1.u[v] = *(const unsigned int*)(p1 + 2 * v);
            }
            s0 = __builtin_amdgcn_wmma_f32_16x16x32_bf16(false, qa[j].v, false, b0.v,
                                                         (short)0, s0, false, false);
            s1 = __builtin_amdgcn_wmma_f32_16x16x32_bf16(false, qa[j].v, false, b1.v,
                                                         (short)0, s1, false, false);
        }

        // ---- online softmax: row max over 32 new cols, rescale, exp, row sum
        float alpha[8];
#pragma unroll
        for (int r = 0; r < 8; ++r) {
            float v = fmaxf(s0[r], s1[r]);
#pragma unroll
            for (int sh = 1; sh < 16; sh <<= 1) v = fmaxf(v, __shfl_xor(v, sh, 32));
            float nm = fmaxf(rowmax[r], v);
            alpha[r] = __expf(rowmax[r] - nm);   // 0 on first iteration
            rowmax[r] = nm;
        }
#pragma unroll
        for (int r = 0; r < 8; ++r) {
            float p0 = __expf(s0[r] - rowmax[r]);
            float p1 = __expf(s1[r] - rowmax[r]);
            s0[r] = p0; s1[r] = p1;
            float rs = p0 + p1;
#pragma unroll
            for (int sh = 1; sh < 16; sh <<= 1) rs += __shfl_xor(rs, sh, 32);
            rowsum[r] = rowsum[r] * alpha[r] + rs;
        }
#pragma unroll
        for (int t2 = 0; t2 < 12; ++t2)
#pragma unroll
            for (int r = 0; r < 8; ++r) acc[t2][r] *= alpha[r];

        // ---- D-layout P (f32) -> A-layout P (bf16) via per-wave LDS bounce
        // Same-wave DS ops execute in order (ISA 7.3): only compiler ordering
        // barriers are needed; no hardware DScnt drain (keeps the pipelined
        // ds_load/wmma schedule intact).
        asm volatile("" ::: "memory");
#pragma unroll
        for (int r = 0; r < 8; ++r) {
            int row = r + 8 * half;
            pb[row * 32 + m]      = f2bf(s0[r]);
            pb[row * 32 + 16 + m] = f2bf(s1[r]);
        }
        asm volatile("" ::: "memory");
        ABf pa;
#pragma unroll
        for (int v = 0; v < 8; ++v) {
            int kb2 = half * 16 + 2 * v;
            pa.u[v] = *(const unsigned int*)(pb + m * 32 + kb2);
        }

        // ---- O += P V : 12 tiles over the 192 channels, K-dim = 32 keys
#pragma unroll
        for (int t2 = 0; t2 < 12; ++t2) {
            ABf bv;
            const unsigned short* vp = vb + (size_t)(t2 * 16 + m) * 32 + half * 16;
#pragma unroll
            for (int v = 0; v < 8; ++v) bv.u[v] = *(const unsigned int*)(vp + 2 * v);
            acc[t2] = __builtin_amdgcn_wmma_f32_16x16x32_bf16(false, pa.v, false, bv.v,
                                                              (short)0, acc[t2], false, false);
        }
    }

    // ---- epilogue: O /= rowsum, scatter ctx^T back to [B][C][N] f32
#pragma unroll
    for (int r = 0; r < 8; ++r) rowsum[r] = 1.0f / rowsum[r];
#pragma unroll
    for (int t2 = 0; t2 < 12; ++t2) {
        int c = t2 * 16 + m;
        float* cp = ctx + ((size_t)b * C_ + c) * N_ + q0;
#pragma unroll
        for (int r = 0; r < 8; ++r) cp[r + 8 * half] = acc[t2][r] * rowsum[r];
    }
}

// ---------------- kernel 3: out = w5 @ ref + b5 + ctx --------------------------
__global__ void k_out(const float* __restrict__ ref, const float* __restrict__ w5,
                      const float* __restrict__ b5, const float* __restrict__ ctx,
                      float* __restrict__ out) {
    __shared__ float wrow[C_];
    int bo = blockIdx.x;
    int b = bo / C_, o = bo % C_;
    int t = threadIdx.x;
    if (t < C_) wrow[t] = w5[o * C_ + t];
    __syncthreads();
    int n = blockIdx.y * blockDim.x + t;
    const float* refb = ref + (size_t)b * C_ * N_ + n;
    float acc = b5[o];
#pragma unroll 8
    for (int c = 0; c < C_; ++c) acc += wrow[c] * refb[(size_t)c * N_];
    size_t oi = ((size_t)b * C_ + o) * N_ + n;
    out[oi] = acc + ctx[oi];
}

extern "C" void kernel_launch(void* const* d_in, const int* in_sizes, int n_in,
                              void* d_out, int out_size, void* d_ws, size_t ws_size,
                              hipStream_t stream) {
    const float* lat = (const float*)d_in[0];
    const float* ref = (const float*)d_in[1];
    const float* w4  = (const float*)d_in[2];
    const float* b4  = (const float*)d_in[3];
    const float* w5  = (const float*)d_in[4];
    const float* b5  = (const float*)d_in[5];
    float* out = (float*)d_out;

    char* ws = (char*)d_ws;
    size_t bf16sz = (size_t)B_ * N_ * C_ * sizeof(unsigned short); // 6 MB each
    unsigned short* nmaj = (unsigned short*)(ws);                  // latp [B][N][C]
    unsigned short* cmaj = (unsigned short*)(ws + bf16sz);         // latp [B][C][N]
    unsigned short* qbf  = (unsigned short*)(ws + 2 * bf16sz);     // refT [B][N][C]
    float*          ctx  = (float*)(ws + 3 * bf16sz);              // [B][C][N] f32

    dim3 g1(B_ * C_, N_ / 256);
    k_latp<<<g1, 256, 0, stream>>>(lat, w4, b4, nmaj, cmaj);
    k_q<<<(B_ * N_ * C_) / 256, 256, 0, stream>>>(ref, qbf);

    dim3 g2(B_, N_ / (16 * WAVES));
    k_attn<<<g2, 32 * WAVES, 0, stream>>>(qbf, nmaj, cmaj, ctx);

    k_out<<<g1, 256, 0, stream>>>(ref, w5, b5, ctx, out);
}
